// QWenAttention_20315195310258
// MI455X (gfx1250) — matrix operand order repaired and checked by reading, not compiled, not run on hardware
//
#include <hip/hip_runtime.h>
#include <math.h>

// ---------------------------------------------------------------------------
// Problem constants (from the reference)
// ---------------------------------------------------------------------------
#define B_    2
#define S_    2048
#define DM_   2048
#define H_    16
#define HD_   128
#define NT_   (B_ * S_)       // 4096 tokens
#define N3DM_ (3 * DM_)       // 6144

typedef __attribute__((ext_vector_type(16))) __bf16 v16bf;
typedef __attribute__((ext_vector_type(8)))  float  v8f;
typedef int i32x4 __attribute__((vector_size(16)));
typedef int i32x2 __attribute__((vector_size(8)));

#define AS1 __attribute__((address_space(1)))
#define AS3 __attribute__((address_space(3)))

// ---------------------------------------------------------------------------
// CDNA5 async global->LDS copies (ASYNCcnt path), feature-guarded so the
// source still compiles on toolchains without the builtins.
// ---------------------------------------------------------------------------
__device__ __forceinline__ void wait_async0() {
#if __has_builtin(__builtin_amdgcn_s_wait_asynccnt)
    __builtin_amdgcn_s_wait_asynccnt(0);
#else
    asm volatile("s_wait_asynccnt 0" ::: "memory");
#endif
}

#if __has_builtin(__builtin_amdgcn_global_load_async_to_lds_b128)
#define HAVE_ASYNC128 1
__device__ __forceinline__ void async_cp16(const void* g, void* l) {
    __builtin_amdgcn_global_load_async_to_lds_b128(
        (AS1 i32x4*)g, (AS3 i32x4*)l, 0, 0);
}
#else
#define HAVE_ASYNC128 0
#endif

#if __has_builtin(__builtin_amdgcn_global_load_async_to_lds_b64)
#define HAVE_ASYNC64 1
__device__ __forceinline__ void async_cp8(const void* g, void* l) {
    __builtin_amdgcn_global_load_async_to_lds_b64(
        (AS1 i32x2*)g, (AS3 i32x2*)l, 0, 0);
}
#else
#define HAVE_ASYNC64 0
#endif

// Assemble a 16-element bf16 fragment from two 16-byte chunks (two ds b128).
__device__ __forceinline__ v16bf frag16(const __bf16* p0, const __bf16* p1) {
    v16bf r;
    ((uint4*)&r)[0] = *(const uint4*)p0;
    ((uint4*)&r)[1] = *(const uint4*)p1;
    return r;
}

__device__ __forceinline__ v8f v8f_zero() {
    v8f z;
#pragma unroll
    for (int i = 0; i < 8; ++i) z[i] = 0.0f;
    return z;
}

// Pack two f32 into packed bf16 dword (v_cvt_pk_bf16_f32 candidate).
__device__ __forceinline__ unsigned pk_bf16(float x, float y) {
    union { __bf16 h[2]; unsigned u; } p;
    p.h[0] = (__bf16)x; p.h[1] = (__bf16)y;
    return p.u;
}

// ---------------------------------------------------------------------------
// Kernel 1/4: C[M,N] = A[M,K] * B[K,N] (+ bias), bf16 WMMA with f32 accum.
// Block = 256 threads = 8 waves; tile 128x128x32; wave grid 4x2, each wave
// owns a 32x64 sub-tile = 2x4 WMMA accumulators.
// ---------------------------------------------------------------------------
template <bool A_BF16, bool BIAS>
__global__ __launch_bounds__(256)
void gemm_bf16_wmma(const void* __restrict__ Ap, const float* __restrict__ Bp,
                    const float* __restrict__ bias, float* __restrict__ Cp,
                    int M, int N, int K)
{
    __shared__ __bf16 As[128][32];   // [m][k]              8 KB
    __shared__ __bf16 Bt[128][32];   // [n][k] (transposed) 8 KB

    const int t    = threadIdx.x;
    const int lane = t & 31;
    const int w    = t >> 5;      // wave 0..7
    const int wm   = w >> 1;      // 0..3
    const int wn   = w & 1;       // 0..1
    const int half = lane >> 4;   // 0/1
    const int l16  = lane & 15;
    const int m0   = blockIdx.y * 128;
    const int n0   = blockIdx.x * 128;

    v8f acc[2][4];
#pragma unroll
    for (int i = 0; i < 2; ++i)
#pragma unroll
        for (int j = 0; j < 4; ++j) acc[i][j] = v8f_zero();

    for (int k0 = 0; k0 < K; k0 += 32) {
        __syncthreads();
        // ---- stage A (128x32) into LDS as bf16 ----
        if (A_BF16) {
            const __bf16* A = (const __bf16*)Ap;
#pragma unroll
            for (int i = 0; i < 4; ++i) {
                int e = t + i * 256;                 // 1024 4-elem groups
                int r = e >> 3, c = (e & 7) << 2;
                const void* src = A + (size_t)(m0 + r) * K + k0 + c;
#if HAVE_ASYNC64
                async_cp8(src, &As[r][c]);           // bf16 pure copy: async DMA
#else
                *(uint2*)&As[r][c] = *(const uint2*)src;
#endif
            }
        } else {
            const float* A = (const float*)Ap;
#pragma unroll
            for (int i = 0; i < 4; ++i) {
                int e = t + i * 256;                 // 1024 float4
                int r = e >> 3, c = (e & 7) << 2;
                float4 v = *(const float4*)(A + (size_t)(m0 + r) * K + k0 + c);
                uint2 pk;
                pk.x = pk_bf16(v.x, v.y);
                pk.y = pk_bf16(v.z, v.w);
                *(uint2*)&As[r][c] = pk;             // one ds_store_b64
            }
        }
        // ---- stage B (32x128) transposed into LDS as bf16 ----
#pragma unroll
        for (int i = 0; i < 4; ++i) {
            int e = t + i * 256;                     // 1024 float4 over 32x128
            int r = e >> 5, c = (e & 31) << 2;       // r = k row, c = n col
            const float* src = Bp + (size_t)(k0 + r) * N + n0 + c;
            float4 v = *(const float4*)src;
            if (i == 0 && k0 + 32 < K)
                __builtin_prefetch(src + (size_t)32 * N, 0, 1);  // next k-tile
            Bt[c + 0][r] = (__bf16)v.x; Bt[c + 1][r] = (__bf16)v.y;
            Bt[c + 2][r] = (__bf16)v.z; Bt[c + 3][r] = (__bf16)v.w;
        }
        if (A_BF16) wait_async0();
        __syncthreads();

        // ---- fragments + 8 WMMAs ----
        v16bf a[2], b[4];
#pragma unroll
        for (int i = 0; i < 2; ++i) {
            int m = wm * 32 + i * 16 + l16;
            a[i] = frag16(&As[m][16 * half], &As[m][16 * half + 8]);
        }
#pragma unroll
        for (int j = 0; j < 4; ++j) {
            int n = wn * 64 + j * 16 + l16;
            b[j] = frag16(&Bt[n][16 * half], &Bt[n][16 * half + 8]);
        }
#pragma unroll
        for (int i = 0; i < 2; ++i)
#pragma unroll
            for (int j = 0; j < 4; ++j)
                acc[i][j] = __builtin_amdgcn_wmma_f32_16x16x32_bf16(
                    false, a[i], false, b[j], (short)0, acc[i][j], false, false);
    }

    // ---- epilogue: C layout row = r + 8*half, col = l16 ----
#pragma unroll
    for (int i = 0; i < 2; ++i)
#pragma unroll
        for (int j = 0; j < 4; ++j) {
            int col = n0 + wn * 64 + j * 16 + l16;
            float bv = BIAS ? bias[col] : 0.0f;
#pragma unroll
            for (int r = 0; r < 8; ++r) {
                int row = m0 + wm * 32 + i * 16 + r + 8 * half;
                Cp[(size_t)row * N + col] = acc[i][j][r] + bv;
            }
        }
}

// ---------------------------------------------------------------------------
// Kernel 2: RoPE (full-rotate, HD=128, pairs d <-> d+-64) + split/convert to
// bf16 Q/K/V laid out [B,H,S,HD] for the attention kernel.
// ---------------------------------------------------------------------------
__global__ __launch_bounds__(128)
void rope_split(const float* __restrict__ qkv, const float* __restrict__ cosb,
                const float* __restrict__ sinb,
                __bf16* __restrict__ Qb, __bf16* __restrict__ Kb,
                __bf16* __restrict__ Vb)
{
    const int token = blockIdx.x;                 // 0..NT-1
    const int b = token / S_, s = token % S_;
    const int d = threadIdx.x;                    // 0..127
    const float c  = cosb[s * HD_ + d];
    const float sn = sinb[s * HD_ + d];
    const int   p  = (d < 64) ? d + 64 : d - 64;
    const float sg = (d < 64) ? -1.0f : 1.0f;
    const float* row = qkv + (size_t)token * N3DM_;
#pragma unroll 4
    for (int h = 0; h < H_; ++h) {
        float q  = row[h * HD_ + d];
        float qp = row[h * HD_ + p];
        float k  = row[DM_ + h * HD_ + d];
        float kp = row[DM_ + h * HD_ + p];
        float v  = row[2 * DM_ + h * HD_ + d];
        size_t o = (((size_t)(b * H_ + h)) * S_ + s) * HD_ + d;
        Qb[o] = (__bf16)(q * c + sg * qp * sn);
        Kb[o] = (__bf16)(k * c + sg * kp * sn);
        Vb[o] = (__bf16)v;
    }
}

// ---------------------------------------------------------------------------
// Kernel 3: causal flash attention.  One block = one (b,h) x 128 q-rows.
// 8 waves, each wave owns 16 q-rows; Q fragments stay in registers for the
// whole KV sweep.  K tile staged with async global->LDS DMA (ASYNCcnt),
// V staged transposed for P.V; P re-laid-out C->A via per-wave LDS.
// ---------------------------------------------------------------------------
__global__ __launch_bounds__(256)
void flash_attn(const __bf16* __restrict__ Qb, const __bf16* __restrict__ Kb,
                const __bf16* __restrict__ Vb, __bf16* __restrict__ ctx)
{
    __shared__ __bf16 Ks[64][HD_];    // 16 KB  [key][d]
    __shared__ __bf16 Vt[HD_][64];    // 16 KB  [d][key]
    __shared__ __bf16 Ps[8][16][64];  // 16 KB  per-wave P tiles

    const int t    = threadIdx.x;
    const int lane = t & 31;
    const int w    = t >> 5;
    const int half = lane >> 4;
    const int l16  = lane & 15;
    const int nqt  = S_ / 128;
    const int bh   = blockIdx.x / nqt;
    const int qb   = (blockIdx.x % nqt) * 128;
    const int b    = bh / H_, h = bh % H_;
    const float scale = 0.08838834764831845f;     // 1/sqrt(128)

    // Q fragments for this wave's 16 rows, all 4 k-steps of d (HD=128).
    const __bf16* Qrow = Qb + ((size_t)bh * S_ + qb + w * 16 + l16) * HD_;
    v16bf qa[4];
#pragma unroll
    for (int ks = 0; ks < 4; ++ks)
        qa[ks] = frag16(Qrow + ks * 32 + 16 * half,
                        Qrow + ks * 32 + 16 * half + 8);

    v8f oacc[8];
#pragma unroll
    for (int db = 0; db < 8; ++db) oacc[db] = v8f_zero();
    float mrow[8], lrow[8];
#pragma unroll
    for (int r = 0; r < 8; ++r) { mrow[r] = -INFINITY; lrow[r] = 0.0f; }
    const int myrow0 = qb + w * 16 + 8 * half;    // + r = this lane's q rows

    const int nkt = qb / 64 + 2;                  // tiles covering keys <= qb+127
    for (int kt = 0; kt < nkt; ++kt) {
        const int kb = kt * 64;
        __syncthreads();
        // ---- stage K tile: async global->LDS DMA (pure bf16 copy) ----
#pragma unroll
        for (int i = 0; i < 4; ++i) {
            int e = t + i * 256;                  // 1024 uint4 over 64x128
            int r = e >> 4, c = (e & 15) << 3;
            const void* src = Kb + ((size_t)bh * S_ + kb + r) * HD_ + c;
#if HAVE_ASYNC128
            async_cp16(src, &Ks[r][c]);
#else
            *(uint4*)&Ks[r][c] = *(const uint4*)src;
#endif
            if (i == 0 && kb + 64 < qb + 128)
                __builtin_prefetch((const char*)src + 64 * HD_ * 2, 0, 1);
        }
        // ---- stage V tile transposed ----
#pragma unroll
        for (int i = 0; i < 4; ++i) {
            int e = t + i * 256;
            int r = e >> 4, c = (e & 15) << 3;
            uint4 u = *(const uint4*)(Vb + ((size_t)bh * S_ + kb + r) * HD_ + c);
            const __bf16* vv = (const __bf16*)&u;
#pragma unroll
            for (int j = 0; j < 8; ++j) Vt[c + j][r] = vv[j];
        }
#if HAVE_ASYNC128
        wait_async0();
#endif
        __syncthreads();

        // ---- S = Q K^T : 4 col-blocks x 4 k-steps = 16 WMMAs ----
        v8f sacc[4];
#pragma unroll
        for (int cb = 0; cb < 4; ++cb) {
            sacc[cb] = v8f_zero();
            const int n = cb * 16 + l16;
#pragma unroll
            for (int ks = 0; ks < 4; ++ks) {
                v16bf kf = frag16(&Ks[n][ks * 32 + 16 * half],
                                  &Ks[n][ks * 32 + 16 * half + 8]);
                sacc[cb] = __builtin_amdgcn_wmma_f32_16x16x32_bf16(
                    false, qa[ks], false, kf, (short)0, sacc[cb], false, false);
            }
        }

        // ---- causal mask + scale + streaming softmax ----
        float rmax[8];
#pragma unroll
        for (int r = 0; r < 8; ++r) rmax[r] = -INFINITY;
#pragma unroll
        for (int cb = 0; cb < 4; ++cb) {
            const int col = kb + cb * 16 + l16;
#pragma unroll
            for (int r = 0; r < 8; ++r) {
                float x = sacc[cb][r] * scale;
                if (col > myrow0 + r) x = -INFINITY;
                sacc[cb][r] = x;
                rmax[r] = fmaxf(rmax[r], x);
            }
        }
#pragma unroll
        for (int off = 1; off < 16; off <<= 1)
#pragma unroll
            for (int r = 0; r < 8; ++r)
                rmax[r] = fmaxf(rmax[r], __shfl_xor(rmax[r], off, 32));

        float alpha[8];
#pragma unroll
        for (int r = 0; r < 8; ++r) {
            float mn = fmaxf(mrow[r], rmax[r]);
            alpha[r] = expf(mrow[r] - mn);        // exp(-inf - finite) = 0
            mrow[r]  = mn;
        }
        float rsum[8];
#pragma unroll
        for (int r = 0; r < 8; ++r) rsum[r] = 0.0f;
#pragma unroll
        for (int cb = 0; cb < 4; ++cb)
#pragma unroll
            for (int r = 0; r < 8; ++r) {
                float p = expf(sacc[cb][r] - mrow[r]);
                sacc[cb][r] = p;
                rsum[r] += p;
            }
#pragma unroll
        for (int off = 1; off < 16; off <<= 1)
#pragma unroll
            for (int r = 0; r < 8; ++r)
                rsum[r] += __shfl_xor(rsum[r], off, 32);
#pragma unroll
        for (int r = 0; r < 8; ++r) lrow[r] = lrow[r] * alpha[r] + rsum[r];
#pragma unroll
        for (int db = 0; db < 8; ++db)
#pragma unroll
            for (int r = 0; r < 8; ++r) oacc[db][r] *= alpha[r];

        // ---- re-layout P (C-layout -> A-layout) via per-wave LDS ----
#pragma unroll
        for (int cb = 0; cb < 4; ++cb)
#pragma unroll
            for (int r = 0; r < 8; ++r)
                Ps[w][r + 8 * half][cb * 16 + l16] = (__bf16)sacc[cb][r];
        // per-wave LDS is in-order; register deps force the dscnt waits.

        // ---- O += P V : 2 k-steps x 8 d-blocks = 16 WMMAs ----
        const __bf16* Pr = &Ps[w][l16][0];
#pragma unroll
        for (int kk = 0; kk < 2; ++kk) {
            v16bf pa = frag16(Pr + kk * 32 + 16 * half,
                              Pr + kk * 32 + 16 * half + 8);
#pragma unroll
            for (int db = 0; db < 8; ++db) {
                v16bf vf = frag16(&Vt[db * 16 + l16][kk * 32 + 16 * half],
                                  &Vt[db * 16 + l16][kk * 32 + 16 * half + 8]);
                oacc[db] = __builtin_amdgcn_wmma_f32_16x16x32_bf16(
                    false, pa, false, vf, (short)0, oacc[db], false, false);
            }
        }
    }

    // ---- normalize + write ctx (bf16, [token][h*HD+d]) ----
#pragma unroll
    for (int r = 0; r < 8; ++r) {
        const float inv = 1.0f / lrow[r];
        const size_t token = (size_t)b * S_ + myrow0 + r;
#pragma unroll
        for (int db = 0; db < 8; ++db) {
            const int col = h * HD_ + db * 16 + l16;
            ctx[token * DM_ + col] = (__bf16)(oacc[db][r] * inv);
        }
    }
}

// ---------------------------------------------------------------------------
// Host launcher
// ---------------------------------------------------------------------------
extern "C" void kernel_launch(void* const* d_in, const int* in_sizes, int n_in,
                              void* d_out, int out_size, void* d_ws, size_t ws_size,
                              hipStream_t stream)
{
    const float* hidden = (const float*)d_in[0];   // [B,S,DM]
    const float* cosb   = (const float*)d_in[1];   // [1,S,1,HD]
    const float* sinb   = (const float*)d_in[2];
    const float* w_attn = (const float*)d_in[3];   // [DM,3DM]
    const float* b_attn = (const float*)d_in[4];   // [3DM]
    const float* w_proj = (const float*)d_in[5];   // [DM,DM]
    float* out = (float*)d_out;                    // [B,S,DM] f32

    char* ws = (char*)d_ws;
    float*  qkv = (float*)ws;                                     // NT*6144 f32
    __bf16* Qb  = (__bf16*)(ws + (size_t)NT_ * N3DM_ * sizeof(float));
    __bf16* Kb  = Qb + (size_t)NT_ * DM_;
    __bf16* Vb  = Kb + (size_t)NT_ * DM_;
    __bf16* ctx = Vb + (size_t)NT_ * DM_;

    // 1) qkv = X @ W_attn + b          (f32 -> bf16 WMMA -> f32)
    gemm_bf16_wmma<false, true><<<dim3(N3DM_ / 128, NT_ / 128), dim3(256), 0, stream>>>(
        (const void*)hidden, w_attn, b_attn, qkv, NT_, N3DM_, DM_);

    // 2) RoPE + split into bf16 Q/K/V [B,H,S,HD]
    rope_split<<<dim3(NT_), dim3(128), 0, stream>>>(qkv, cosb, sinb, Qb, Kb, Vb);

    // 3) causal flash attention -> ctx (bf16, [token][DM])
    flash_attn<<<dim3(B_ * H_ * (S_ / 128)), dim3(256), 0, stream>>>(Qb, Kb, Vb, ctx);

    // 4) out = ctx @ W_proj            (bf16 A, f32 out)
    gemm_bf16_wmma<true, false><<<dim3(DM_ / 128, NT_ / 128), dim3(256), 0, stream>>>(
        (const void*)ctx, w_proj, nullptr, out, NT_, DM_, DM_);
}
